// PointNetFPModule_4148938408195
// MI455X (gfx1250) — matrix operand order repaired and checked by reading, not compiled, not run on hardware
//
#include <hip/hip_runtime.h>
#include <hip/hip_bf16.h>

// ---------------- problem constants (from setup_inputs) ----------------
#define B_    8
#define N_    8192
#define M_    2048
#define C_CTR 256
#define C_PTS 128
#define C_IN  384    // C_CTR + C_PTS
#define C_O   256    // C1 == C2
#define D_    64
#define GN_G  8

typedef _Float16 v16h  __attribute__((ext_vector_type(16)));
typedef float    v8f   __attribute__((ext_vector_type(8)));
typedef unsigned uint4v __attribute__((ext_vector_type(4)));
typedef int      i4v    __attribute__((ext_vector_type(4)));

union AFrag { v16h v; _Float16 h[16]; unsigned u[8]; uint4v q[2]; };

// gfx1250 async global->LDS DMA path (guarded: falls back to manual copy).
// Probe-confirmed signature: (int4 addrspace(1)*, int4 addrspace(3)*, imm offset, imm cpol)
#if defined(__AMDGCN__) && __has_builtin(__builtin_amdgcn_global_load_async_to_lds_b128) && \
    __has_builtin(__builtin_amdgcn_s_wait_asynccnt)
#define USE_ASYNC_LDS 1
typedef __attribute__((address_space(1))) i4v* g_i4p;   // global int4*
typedef __attribute__((address_space(3))) i4v* l_i4p;   // LDS int4*
#else
#define USE_ASYNC_LDS 0
#endif

// ---------------- f32 -> f16 conversion ----------------
__global__ void f32_to_f16_kernel(const float* __restrict__ src,
                                  _Float16* __restrict__ dst, int n) {
  int i = blockIdx.x * blockDim.x + threadIdx.x;
  if (i < n) dst[i] = (_Float16)src[i];
}

// ---------------- 3-NN + weighted feature interpolation ----------------
// One thread per point. Centers' coords staged in LDS (24 KB).
// Writes interp features (f16) into rows [0,256) of the concat buffer.
__global__ void __launch_bounds__(256)
knn_interp_kernel(const float* __restrict__ pc, const float* __restrict__ cc,
                  const float* __restrict__ cf, _Float16* __restrict__ xcat) {
  __shared__ __align__(16) float cx[M_];
  __shared__ __align__(16) float cy[M_];
  __shared__ __align__(16) float cz[M_];
  const int b = blockIdx.y;
  const int n = blockIdx.x * 256 + threadIdx.x;
  const float* ccb = cc + (size_t)b * 3 * M_;
  for (int i = threadIdx.x; i < M_; i += 256) {
    cx[i] = ccb[i];
    cy[i] = ccb[M_ + i];
    cz[i] = ccb[2 * M_ + i];
  }
  __syncthreads();

  const float px = pc[((size_t)b * 3 + 0) * N_ + n];
  const float py = pc[((size_t)b * 3 + 1) * N_ + n];
  const float pz = pc[((size_t)b * 3 + 2) * N_ + n];

  float d0 = 3.4e38f, d1 = 3.4e38f, d2 = 3.4e38f;
  int   i0 = 0, i1 = 0, i2 = 0;
  for (int m = 0; m < M_; ++m) {
    float dx = px - cx[m], dy = py - cy[m], dz = pz - cz[m];
    float d = fmaf(dx, dx, fmaf(dy, dy, dz * dz));
    if (d < d2) {
      if (d < d1) {
        d2 = d1; i2 = i1;
        if (d < d0) { d1 = d0; i1 = i0; d0 = d; i0 = m; }
        else        { d1 = d;  i1 = m; }
      } else { d2 = d; i2 = m; }
    }
  }
  float w0 = 1.0f / (d0 + 1e-8f);
  float w1 = 1.0f / (d1 + 1e-8f);
  float w2 = 1.0f / (d2 + 1e-8f);
  const float ws = 1.0f / (w0 + w1 + w2);
  w0 *= ws; w1 *= ws; w2 *= ws;

  const float* cfb = cf + (size_t)b * C_CTR * M_;
  _Float16* out = xcat + (size_t)b * C_IN * N_ + n;
  for (int c = 0; c < C_CTR; ++c) {
    const float* row = cfb + (size_t)c * M_;
    float v = w0 * row[i0] + w1 * row[i1] + w2 * row[i2];
    out[(size_t)c * N_] = (_Float16)v;
  }
}

// ---------------- points_features -> concat rows [256,384) as f16 ----------------
__global__ void pf_to_xcat_kernel(const float* __restrict__ pf,
                                  _Float16* __restrict__ xcat) {
  long i = (long)blockIdx.x * 256 + threadIdx.x;
  const long per_b = (long)C_PTS * N_;
  int  b = (int)(i / per_b);
  long r = i - (long)b * per_b;
  int  c = (int)(r / N_);
  int  n = (int)(r - (long)c * N_);
  xcat[((size_t)b * C_IN + C_CTR + c) * N_ + n] = (_Float16)pf[i];
}

// ---------------- WMMA GEMM: Y[b][o][n] = sum_k Wh[o][k]*Xh[b][k][n] + bias[o] ----------------
// Block tile 64(M) x 32(N), 8 waves as 4x2 grid of 16x16 WMMA tiles, K stepped by 32.
// A is staged row-major (async global->LDS DMA when available); B is staged
// TRANSPOSED (Bst[col][k]) so both fragments load as 2x ds_load_b128 per wave.
// Fragment layouts per ISA 7.12.2:
//   A (16-bit 16x32): lane half selects K-octet; VGPR v<4 -> K=8*half+2v{,+1},
//                     v>=4 -> K=16+8*half+2(v-4){,+1}; M = lane%16.
//   B (16-bit 32x16): VGPR v -> same K mapping as A (K-octet per half); N = lane%16.
//   C/D (f32 16x16):  VGPR r -> M = r + 8*half; N = lane%16.
template<int K>
__global__ void __launch_bounds__(256)
wmma_gemm_kernel(const _Float16* __restrict__ Wh, const _Float16* __restrict__ Xh,
                 const float* __restrict__ bias, float* __restrict__ Y) {
  constexpr int ASTR = 40;   // halfs; 80 B rows: 16B-aligned, bank-spread (gcd(20,64)=4)
  constexpr int BSTR = 40;
  __shared__ __align__(16) _Float16 As[64 * ASTR];     // 5.0 KB, [row][k]
  __shared__ __align__(16) _Float16 Bst[32 * BSTR];    // 2.5 KB, [col][k] (transposed)

  const int tid   = threadIdx.x;
  const int lane  = tid & 31;
  const int wave  = tid >> 5;      // 0..7
  const int wm    = wave & 3;      // M tile of wave
  const int wn    = wave >> 2;     // N tile of wave
  const int half_ = lane >> 4;
  const int ml    = lane & 15;

  const int b  = blockIdx.z;
  const int n0 = blockIdx.x * 32;
  const int o0 = blockIdx.y * 64;
  const _Float16* Xb = Xh + (size_t)b * K * N_;

  v8f acc = {0.f, 0.f, 0.f, 0.f, 0.f, 0.f, 0.f, 0.f};
  constexpr int KT = K / 32;

  // A-staging geometry: one 16B chunk per thread (64 rows x 4 chunks)
  const int arow = tid >> 2;        // 0..63
  const int achk = (tid & 3) * 8;   // half offset within the 32-half row

  for (int kt = 0; kt < KT; ++kt) {
    const int k0 = kt * 32;
    __syncthreads();   // previous tile fully consumed

    // ---- stage A: 64 rows x 32 halfs ----
#if USE_ASYNC_LDS
    __builtin_amdgcn_global_load_async_to_lds_b128(
        (g_i4p)(Wh + (size_t)(o0 + arow) * K + k0 + achk),
        (l_i4p)(&As[arow * ASTR + achk]), 0, 0);
#else
    *(uint4v*)&As[arow * ASTR + achk] =
        *(const uint4v*)&Wh[(size_t)(o0 + arow) * K + k0 + achk];
#endif

    // ---- stage B transposed: global [k][n] -> LDS [n][k] ----
    unsigned short* BstU = (unsigned short*)Bst;
#pragma unroll
    for (int s = 0; s < 2; ++s) {
      int u  = tid + 256 * s;       // 0..511
      int kr = u >> 4;              // k within tile (0..31)
      int j  = u & 15;              // n-pair within tile
      unsigned w = *(const unsigned*)&Xb[(size_t)(k0 + kr) * N_ + n0 + 2 * j];
      BstU[(2 * j) * BSTR + kr]     = (unsigned short)(w & 0xffffu);
      BstU[(2 * j + 1) * BSTR + kr] = (unsigned short)(w >> 16);
    }
    if (kt + 1 < KT) {
      // hint next K tile of the activation stream into cache
      __builtin_prefetch(&Xb[(size_t)(k0 + 32) * N_ + n0 + 2 * lane], 0, 1);
    }
#if USE_ASYNC_LDS
    __builtin_amdgcn_s_wait_asynccnt(0);
#endif
    __syncthreads();

    // ---- fragments: 2x ds_load_b128 each ----
    AFrag a, bf;
    const int abase = (wm * 16 + ml) * ASTR;
    a.q[0] = *(const uint4v*)&As[abase + 8 * half_];        // K = 8h .. 8h+7
    a.q[1] = *(const uint4v*)&As[abase + 16 + 8 * half_];   // K = 16+8h .. 16+8h+7
    const int bbase = (wn * 16 + ml) * BSTR + 8 * half_;
    bf.q[0] = *(const uint4v*)&Bst[bbase];                  // K = 8h .. 8h+7
    bf.q[1] = *(const uint4v*)&Bst[bbase + 16];             // K = 16+8h .. 16+8h+7

    acc = __builtin_amdgcn_wmma_f32_16x16x32_f16(
        false, a.v, false, bf.v, (short)0, acc, false, false);
  }

  // epilogue: fuse bias, write f32
  float* Yb = Y + (size_t)b * C_O * N_;
  const int ncol = n0 + wn * 16 + ml;
#pragma unroll
  for (int r = 0; r < 8; ++r) {
    int oo = o0 + wm * 16 + r + 8 * half_;
    Yb[(size_t)oo * N_ + ncol] = acc[r] + bias[oo];
  }
}

// ---------------- GroupNorm stats: one block per (group, batch) ----------------
__global__ void __launch_bounds__(256)
gn_stats_kernel(const float* __restrict__ Y, float* __restrict__ stats) {
  __shared__ float s1[256];
  __shared__ float s2[256];
  const int g = blockIdx.x, b = blockIdx.y;
  const float* base = Y + ((size_t)b * C_O + g * (C_O / GN_G)) * N_;
  const long cnt = (long)(C_O / GN_G) * N_;   // contiguous region
  float sum = 0.f, ss = 0.f;
  for (long i = threadIdx.x; i < cnt; i += 256) {
    float v = base[i];
    sum += v;
    ss  = fmaf(v, v, ss);
  }
  s1[threadIdx.x] = sum;
  s2[threadIdx.x] = ss;
  __syncthreads();
  for (int off = 128; off > 0; off >>= 1) {
    if (threadIdx.x < off) {
      s1[threadIdx.x] += s1[threadIdx.x + off];
      s2[threadIdx.x] += s2[threadIdx.x + off];
    }
    __syncthreads();
  }
  if (threadIdx.x == 0) {
    float mean = s1[0] / (float)cnt;
    float var  = s2[0] / (float)cnt - mean * mean;
    stats[(b * GN_G + g) * 2 + 0] = mean;
    stats[(b * GN_G + g) * 2 + 1] = rsqrtf(var + 1e-5f);
  }
}

// ---------------- fused GroupNorm-affine + Swish, templated output dtype ----------------
template<typename OUT>
__global__ void gn_swish_kernel(const float* __restrict__ Y, const float* __restrict__ stats,
                                const float* __restrict__ gw, const float* __restrict__ gb,
                                OUT* __restrict__ out) {
  long i = (long)blockIdx.x * 256 + threadIdx.x;
  int b = (int)(i / ((long)C_O * N_));
  int c = (int)((i / N_) & (C_O - 1));
  int g = c >> 5;
  float mean = stats[(b * GN_G + g) * 2 + 0];
  float rstd = stats[(b * GN_G + g) * 2 + 1];
  float t = (Y[i] - mean) * rstd * gw[c] + gb[c];
  float s = t / (1.0f + __expf(-t));   // swish = t * sigmoid(t)
  out[i] = (OUT)s;
}

// ---------------- output tail: coords copy + time-embedding broadcast ----------------
__global__ void copy_f32_kernel(const float* __restrict__ src, float* __restrict__ dst, int n) {
  int i = blockIdx.x * 256 + threadIdx.x;
  if (i < n) dst[i] = src[i];
}

__global__ void time_bcast_kernel(const float* __restrict__ te, float* __restrict__ dst) {
  long i = (long)blockIdx.x * 256 + threadIdx.x;
  int b = (int)(i / ((long)D_ * N_));
  int d = (int)((i / N_) & (D_ - 1));
  dst[i] = te[b * D_ + d];
}

// ---------------- host-side orchestration ----------------
extern "C" void kernel_launch(void* const* d_in, const int* in_sizes, int n_in,
                              void* d_out, int out_size, void* d_ws, size_t ws_size,
                              hipStream_t stream) {
  const float* pc  = (const float*)d_in[0];   // points_coords   [B,3,N]
  const float* cc  = (const float*)d_in[1];   // centers_coords  [B,3,M]
  const float* cf  = (const float*)d_in[2];   // centers_features[B,256,M]
  const float* pf  = (const float*)d_in[3];   // points_features [B,128,N]
  const float* te  = (const float*)d_in[4];   // time_emb        [B,64,1]
  const float* W1  = (const float*)d_in[5];
  const float* b1  = (const float*)d_in[6];
  const float* g1w = (const float*)d_in[7];
  const float* g1b = (const float*)d_in[8];
  const float* W2  = (const float*)d_in[9];
  const float* b2  = (const float*)d_in[10];
  const float* g2w = (const float*)d_in[11];
  const float* g2b = (const float*)d_in[12];

  // workspace layout (all region sizes are multiples of 256 B)
  char* ws = (char*)d_ws;
  size_t off = 0;
  _Float16* xcat = (_Float16*)(ws + off); off += (size_t)B_ * C_IN * N_ * sizeof(_Float16); // 50.3 MB
  float*    y    = (float*)(ws + off);    off += (size_t)B_ * C_O * N_ * sizeof(float);     // 67.1 MB
  _Float16* w1h  = (_Float16*)(ws + off); off += (size_t)C_O * C_IN * sizeof(_Float16);
  _Float16* w2h  = (_Float16*)(ws + off); off += (size_t)C_O * C_O * sizeof(_Float16);
  float* stats1  = (float*)(ws + off);    off += (size_t)B_ * GN_G * 2 * sizeof(float);
  float* stats2  = (float*)(ws + off);    off += (size_t)B_ * GN_G * 2 * sizeof(float);
  _Float16* x1h  = xcat;  // overlay: concat buffer is dead after GEMM1

  // weights to f16
  f32_to_f16_kernel<<<(C_O * C_IN + 255) / 256, 256, 0, stream>>>(W1, w1h, C_O * C_IN);
  f32_to_f16_kernel<<<(C_O * C_O + 255) / 256, 256, 0, stream>>>(W2, w2h, C_O * C_O);

  // concat buffer: interp (rows 0..255) + points_features (rows 256..383), f16
  knn_interp_kernel<<<dim3(N_ / 256, B_), 256, 0, stream>>>(pc, cc, cf, xcat);
  pf_to_xcat_kernel<<<(int)(((long)B_ * C_PTS * N_) / 256), 256, 0, stream>>>(pf, xcat);

  // layer 1: GEMM + GN stats + fused norm/swish (f16 out for GEMM2)
  wmma_gemm_kernel<C_IN><<<dim3(N_ / 32, C_O / 64, B_), 256, 0, stream>>>(w1h, xcat, b1, y);
  gn_stats_kernel<<<dim3(GN_G, B_), 256, 0, stream>>>(y, stats1);
  gn_swish_kernel<_Float16><<<(int)(((long)B_ * C_O * N_) / 256), 256, 0, stream>>>(
      y, stats1, g1w, g1b, x1h);

  // layer 2: GEMM + GN stats + fused norm/swish (f32 straight into d_out)
  wmma_gemm_kernel<C_O><<<dim3(N_ / 32, C_O / 64, B_), 256, 0, stream>>>(w2h, x1h, b2, y);
  gn_stats_kernel<<<dim3(GN_G, B_), 256, 0, stream>>>(y, stats2);

  float* out_x = (float*)d_out;
  gn_swish_kernel<float><<<(int)(((long)B_ * C_O * N_) / 256), 256, 0, stream>>>(
      y, stats2, g2w, g2b, out_x);

  // remaining tuple outputs
  float* out_pc = out_x + (size_t)B_ * C_O * N_;
  copy_f32_kernel<<<(B_ * 3 * N_ + 255) / 256, 256, 0, stream>>>(pc, out_pc, B_ * 3 * N_);
  float* out_te = out_pc + (size_t)B_ * 3 * N_;
  time_bcast_kernel<<<(int)(((long)B_ * D_ * N_) / 256), 256, 0, stream>>>(te, out_te);
}
